// QANetCore_48017734369913
// MI455X (gfx1250) — matrix operand order repaired and checked by reading, not compiled, not run on hardware
//
#include <hip/hip_runtime.h>
#include <math.h>

// ---------------------------------------------------------------------------
// QANet core forward for MI455X (gfx1250): all GEMMs via v_wmma_f32_16x16x32_bf16
// Fragment-order LDS staging + double buffering + b128 LDS fragment loads.
// k_gemm templated on transA/transB so staging is straight-line code.
// ---------------------------------------------------------------------------

typedef __attribute__((ext_vector_type(16))) __bf16          v16bf;
typedef __attribute__((ext_vector_type(16))) unsigned short  v16u;
typedef __attribute__((ext_vector_type(8)))  unsigned short  v8u;
typedef __attribute__((ext_vector_type(8)))  float           v8f;

// compile-time boolean tags for the staging fast/slow path
struct TrueT  { static constexpr bool value = true;  };
struct FalseT { static constexpr bool value = false; };

__device__ __forceinline__ unsigned short f2bf(float f) {
  unsigned u = __float_as_uint(f);
  u += 0x7FFFu + ((u >> 16) & 1u);   // round-to-nearest-even
  return (unsigned short)(u >> 16);
}

struct GemmP {
  const float* A; const float* B; float* C;
  const float* bias; const float* resid;
  int M, N, K;
  int lda, ldb, ldc, ldres;
  int transA, transB, relu, zshift;   // batch z -> (z>>zshift, z&mask)
  long sA0, sA1, sB0, sB1, sC0, sC1, sR0, sR1;
  float alpha;
};

// 64x64 output tile per workgroup (256 threads = 8 waves, wave32).
// Each wave computes a 32x16 sub-tile: 2x v_wmma_f32_16x16x32_bf16 per K-step.
// LDS tiles are stored in WMMA fragment order:
//   A frag block fb (16 rows): lane l holds row fb*16+(l&15);
//     element e -> K = (e&7) + 8*(l>>4) + 16*(e>>3)
//   B frag block cb (16 cols): lane l holds col cb*16+(l&15);
//     element e -> K = (e&7) + 8*(l>>4) + 16*(e>>3)
template <int TA, int TB>
__global__ __launch_bounds__(256) void k_gemm(GemmP p) {
  __shared__ alignas(32) unsigned short sAf[2][4][32][16];  // [buf][rowblk][lane][e]
  __shared__ alignas(32) unsigned short sBf[2][4][32][16];  // [buf][colblk][lane][e]

  int z  = blockIdx.z;
  int z0 = z >> p.zshift;
  int z1 = z & ((1 << p.zshift) - 1);
  const float* Ap = p.A + z0 * p.sA0 + z1 * p.sA1;
  const float* Bp = p.B + z0 * p.sB0 + z1 * p.sB1;
  float*       Cp = p.C + z0 * p.sC0 + z1 * p.sC1;
  const float* Rp = p.resid ? (p.resid + z0 * p.sR0 + z1 * p.sR1) : (const float*)0;

  int m0 = blockIdx.y * 64;
  int n0 = blockIdx.x * 64;
  int tid = threadIdx.x;
  int lane = tid & 31;
  int wv = tid >> 5;
  int wm = wv >> 2;            // 0..1 : 32-row band
  int wn = wv & 3;             // 0..3 : 16-col band

  // staging decomposition: thread owns 8 consecutive fragment elements
  int eg    = tid & 1;          // e-group: 0 -> e 0..7, 1 -> e 8..15
  int slane = (tid >> 1) & 31;  // fragment lane
  int sblk  = (tid >> 6) & 3;   // fragment block
  int a_m  = sblk * 16 + (slane & 15);
  int a_kb = eg * 16 + 8 * (slane >> 4);           // K base, +j (j=0..7) contiguous
  int b_n  = sblk * 16 + (slane & 15);
  int b_kb = eg * 16 + 8 * (slane >> 4);           // K base, +j contiguous

  int gmA = m0 + a_m;
  int gnB = n0 + b_n;

  // K-invariant base addresses (per trans mode)
  const float* aBase = TA ? (Ap + gmA) : (Ap + (long)gmA * p.lda);
  const float* bBase = TB ? (Bp + (long)gnB * p.ldb) : (Bp + gnB);

  bool fullMN = (m0 + 64 <= p.M) && (n0 + 64 <= p.N);

  // stage one K-step (32 wide) of both tiles into buffer `buf`, fragment order
  auto stage = [&](int buf, int k0, auto FULLC) {
    constexpr bool FULL = decltype(FULLC)::value;
    int kA = k0 + a_kb;
    int kB = k0 + b_kb;
    v8u av, bv;
    if (!TA) {
      const float* src = aBase + kA;
#pragma unroll
      for (int j = 0; j < 8; ++j) {
        float v = 0.f;
        if (FULL || (gmA < p.M && kA + j < p.K)) v = src[j];
        av[j] = f2bf(v);
      }
    } else {
      const float* src = aBase + (long)kA * p.lda;
#pragma unroll
      for (int j = 0; j < 8; ++j) {
        float v = 0.f;
        if (FULL || (gmA < p.M && kA + j < p.K)) v = src[(long)j * p.lda];
        av[j] = f2bf(v);
      }
    }
    if (!TB) {
      const float* src = bBase + (long)kB * p.ldb;
#pragma unroll
      for (int j = 0; j < 8; ++j) {
        float v = 0.f;
        if (FULL || (gnB < p.N && kB + j < p.K)) v = src[(long)j * p.ldb];
        bv[j] = f2bf(v);
      }
    } else {
      const float* src = bBase + kB;
#pragma unroll
      for (int j = 0; j < 8; ++j) {
        float v = 0.f;
        if (FULL || (gnB < p.N && kB + j < p.K)) v = src[j];
        bv[j] = f2bf(v);
      }
    }
    *(v8u*)&sAf[buf][sblk][slane][eg * 8] = av;   // one ds_store_b128
    *(v8u*)&sBf[buf][sblk][slane][eg * 8] = bv;   // one ds_store_b128
  };

  v8f c0 = {0.f,0.f,0.f,0.f,0.f,0.f,0.f,0.f};
  v8f c1 = {0.f,0.f,0.f,0.f,0.f,0.f,0.f,0.f};

  int buf = 0;
  if (fullMN && 32 <= p.K) stage(0, 0, TrueT{});
  else                     stage(0, 0, FalseT{});
  __syncthreads();

  for (int k0 = 0; k0 < p.K; k0 += 32) {
    // fragment loads: 32B contiguous per lane -> 2x ds_load_b128 each
    v16u au0 = *(const v16u*)&sAf[buf][wm * 2][lane][0];
    v16u au1 = *(const v16u*)&sAf[buf][wm * 2 + 1][lane][0];
    v16u bu  = *(const v16u*)&sBf[buf][wn][lane][0];

    // prefetch the k0+64 tile while computing (gfx1250 global_prefetch_b8)
    if (fullMN && k0 + 96 <= p.K) {
      const float* pa = TA ? aBase + (long)(k0 + 64 + a_kb) * p.lda
                           : aBase + (k0 + 64 + a_kb);
      const float* pb = TB ? bBase + (k0 + 64 + b_kb)
                           : bBase + (long)(k0 + 64 + b_kb) * p.ldb;
      __builtin_prefetch(pa, 0, 1);
      __builtin_prefetch(pb, 0, 1);
    }

    // stage next K-step into the other buffer while WMMAs run
    int kn = k0 + 32;
    if (kn < p.K) {
      if (fullMN && kn + 32 <= p.K) stage(buf ^ 1, kn, TrueT{});
      else                          stage(buf ^ 1, kn, FalseT{});
    }

    v16bf a0 = __builtin_bit_cast(v16bf, au0);
    v16bf a1 = __builtin_bit_cast(v16bf, au1);
    v16bf bb = __builtin_bit_cast(v16bf, bu);
    c0 = __builtin_amdgcn_wmma_f32_16x16x32_bf16(false, a0, false, bb, (short)0, c0, false, false);
    c1 = __builtin_amdgcn_wmma_f32_16x16x32_bf16(false, a1, false, bb, (short)0, c1, false, false);

    __syncthreads();
    buf ^= 1;
  }

  // epilogue: alpha, +bias[m], relu, +residual
  int ng = n0 + wn * 16 + (lane & 15);
  int mrow = (lane >> 4) * 8;
  if (ng < p.N) {
#pragma unroll
    for (int r = 0; r < 8; ++r) {
      int mg = m0 + wm * 32 + mrow + r;
      if (mg < p.M) {
        float t = p.alpha * c0[r];
        if (p.bias) t += p.bias[mg];
        if (p.relu) t = fmaxf(t, 0.f);
        if (Rp) t += Rp[(long)mg * p.ldres + ng];
        Cp[(long)mg * p.ldc + ng] = t;
      }
      int mg1 = m0 + wm * 32 + mrow + r + 16;
      if (mg1 < p.M) {
        float t = p.alpha * c1[r];
        if (p.bias) t += p.bias[mg1];
        if (p.relu) t = fmaxf(t, 0.f);
        if (Rp) t += Rp[(long)mg1 * p.ldres + ng];
        Cp[(long)mg1 * p.ldc + ng] = t;
      }
    }
  }
}

// -------------------------- elementwise / reductions ------------------------

// dst = src + tensor2tensor positional signal, x layout [B,128,L]
__global__ void k_posadd(float* dst, const float* src, int L, int total) {
  int i = blockIdx.x * 256 + threadIdx.x;
  if (i >= total) return;
  int l = i % L;
  int c = (i / L) & 127;
  const float INC = 0.1461477f;   // ln(1e4)/63
  float a = (float)l * __expf(-(float)(c & 63) * INC);
  dst[i] = src[i] + ((c < 64) ? __sinf(a) : __cosf(a));
}

// LayerNorm over channel dim (128) of [B,128,L]; one wave per (b,l)
__global__ __launch_bounds__(256) void k_lnc(const float* x, float* y,
                                             const float* g, const float* bt,
                                             int L, int BL) {
  int wv = threadIdx.x >> 5, lane = threadIdx.x & 31;
  int pos = blockIdx.x * 8 + wv;
  if (pos >= BL) return;
  int b = pos / L, l = pos % L;
  const float* xp = x + (size_t)b * 128 * L + l;
  float v0 = xp[(size_t)lane * L];
  float v1 = xp[(size_t)(lane + 32) * L];
  float v2 = xp[(size_t)(lane + 64) * L];
  float v3 = xp[(size_t)(lane + 96) * L];
  float sm = v0 + v1 + v2 + v3;
  for (int o = 16; o; o >>= 1) sm += __shfl_xor(sm, o, 32);
  float mu = sm * (1.f / 128.f);
  float d0 = v0 - mu, d1 = v1 - mu, d2 = v2 - mu, d3 = v3 - mu;
  float vs = d0 * d0 + d1 * d1 + d2 * d2 + d3 * d3;
  for (int o = 16; o; o >>= 1) vs += __shfl_xor(vs, o, 32);
  float rs = rsqrtf(vs * (1.f / 128.f) + 1e-5f);
  float* yp = y + (size_t)b * 128 * L + l;
  yp[(size_t)lane * L]        = d0 * rs * g[lane]      + bt[lane];
  yp[(size_t)(lane + 32) * L] = d1 * rs * g[lane + 32] + bt[lane + 32];
  yp[(size_t)(lane + 64) * L] = d2 * rs * g[lane + 64] + bt[lane + 64];
  yp[(size_t)(lane + 96) * L] = d3 * rs * g[lane + 96] + bt[lane + 96];
}

// depthwise conv, "same" padding, x/y [B,128,L], w [128,k]
__global__ void k_dwconv(const float* x, float* y, const float* w,
                         int ksz, int L, int total) {
  int i = blockIdx.x * 256 + threadIdx.x;
  if (i >= total) return;
  int l = i % L;
  int c = (i / L) & 127;
  const float* row = x + (i - l);
  const float* wc = w + (size_t)c * ksz;
  int pad = ksz >> 1;
  float acc = 0.f;
  for (int t = 0; t < ksz; ++t) {
    int ll = l + t - pad;
    if (ll >= 0 && ll < L) acc += row[ll] * wc[t];
  }
  y[i] = acc;
}

// masked softmax over last dim of logits [B*H, Lq, Lk]; mask [B, Lk]
__global__ __launch_bounds__(256) void k_attn_softmax(float* lg, const float* mask,
                                                      int Hh, int Lq, int Lk) {
  int q = blockIdx.x;
  int z = blockIdx.y;
  int b = z / Hh;
  float* row = lg + ((size_t)z * Lq + q) * (size_t)Lk;
  const float* mk = mask + (size_t)b * Lk;
  __shared__ float sred[8];
  int tid = threadIdx.x, lane = tid & 31, wv = tid >> 5;
  float mx = -3.4e38f;
  for (int k = tid; k < Lk; k += 256) {
    float m = mk[k];
    float v = row[k] * m + (1.f - m) * (-1e30f);
    row[k] = v;
    mx = fmaxf(mx, v);
  }
  for (int o = 16; o; o >>= 1) mx = fmaxf(mx, __shfl_xor(mx, o, 32));
  if (lane == 0) sred[wv] = mx;
  __syncthreads();
  mx = sred[0];
#pragma unroll
  for (int w = 1; w < 8; ++w) mx = fmaxf(mx, sred[w]);
  __syncthreads();
  float sm = 0.f;
  for (int k = tid; k < Lk; k += 256) {
    float e = __expf(row[k] - mx);
    row[k] = e;
    sm += e;
  }
  for (int o = 16; o; o >>= 1) sm += __shfl_xor(sm, o, 32);
  if (lane == 0) sred[wv] = sm;
  __syncthreads();
  sm = 0.f;
#pragma unroll
  for (int w = 0; w < 8; ++w) sm += sred[w];
  float inv = 1.f / sm;
  for (int k = tid; k < Lk; k += 256) row[k] *= inv;
}

// softmax over q (last dim, Lq==blockDim==64) of S [B,Lc,Lq] with qmask [B,Lq]
__global__ __launch_bounds__(64) void k_cq_softmax_q(const float* S, float* S1,
                                                     const float* qmask, int Lc, int Lq) {
  int rid = blockIdx.x;          // b*Lc + l
  int b = rid / Lc;
  const float* row = S + (size_t)rid * Lq;
  float* orow = S1 + (size_t)rid * Lq;
  int tid = threadIdx.x;
  __shared__ float sr[2];
  float m = qmask[(size_t)b * Lq + tid];
  float v = row[tid] * m + (1.f - m) * (-1e30f);
  float mx = v;
  for (int o = 16; o; o >>= 1) mx = fmaxf(mx, __shfl_xor(mx, o, 32));
  if ((tid & 31) == 0) sr[tid >> 5] = mx;
  __syncthreads();
  mx = fmaxf(sr[0], sr[1]);
  __syncthreads();
  float e = __expf(v - mx);
  float sm = e;
  for (int o = 16; o; o >>= 1) sm += __shfl_xor(sm, o, 32);
  if ((tid & 31) == 0) sr[tid >> 5] = sm;
  __syncthreads();
  sm = sr[0] + sr[1];
  orow[tid] = e / sm;
}

// softmax over l (Lc==512, stride Lq) of S for each (b,q), with cmask [B,Lc]
__global__ __launch_bounds__(256) void k_cq_softmax_c(const float* S, float* S2,
                                                      const float* cmask, int Lc, int Lq) {
  int z = blockIdx.x;            // b*Lq + q
  int b = z / Lq, q = z % Lq;
  const float* base = S + (size_t)b * Lc * Lq + q;
  float* obase = S2 + (size_t)b * Lc * Lq + q;
  const float* mk = cmask + (size_t)b * Lc;
  __shared__ float sred[8];
  int tid = threadIdx.x, lane = tid & 31, wv = tid >> 5;
  float vals[2];
  float mx = -3.4e38f;
#pragma unroll
  for (int it = 0; it < 2; ++it) {     // Lc = 512 = 2*256
    int l = tid + it * 256;
    float m = mk[l];
    float v = base[(size_t)l * Lq] * m + (1.f - m) * (-1e30f);
    vals[it] = v;
    mx = fmaxf(mx, v);
  }
  for (int o = 16; o; o >>= 1) mx = fmaxf(mx, __shfl_xor(mx, o, 32));
  if (lane == 0) sred[wv] = mx;
  __syncthreads();
  mx = sred[0];
#pragma unroll
  for (int w = 1; w < 8; ++w) mx = fmaxf(mx, sred[w]);
  __syncthreads();
  float sm = 0.f;
#pragma unroll
  for (int it = 0; it < 2; ++it) {
    vals[it] = __expf(vals[it] - mx);
    sm += vals[it];
  }
  for (int o = 16; o; o >>= 1) sm += __shfl_xor(sm, o, 32);
  if (lane == 0) sred[wv] = sm;
  __syncthreads();
  sm = 0.f;
#pragma unroll
  for (int w = 0; w < 8; ++w) sm += sred[w];
  float inv = 1.f / sm;
#pragma unroll
  for (int it = 0; it < 2; ++it) {
    int l = tid + it * 256;
    obase[(size_t)l * Lq] = vals[it] * inv;
  }
}

// dst[b][rowOff+r][l] = src[b][r][l]  (channel concat)
__global__ void k_concat_rows(float* dst, const float* src, int rows, int L,
                              int rowOff, int rowsTotal, int total) {
  int i = blockIdx.x * 256 + threadIdx.x;
  if (i >= total) return;
  int l = i % L;
  int r = (i / L) % rows;
  int b = i / (L * rows);
  dst[((size_t)b * rowsTotal + rowOff + r) * (size_t)L + l] = src[i];
}

// dst[b,c,l] = src[b,c,l] * w[c]
__global__ void k_scale_rows(float* dst, const float* src, const float* w,
                             int C, int L, int total) {
  int i = blockIdx.x * 256 + threadIdx.x;
  if (i >= total) return;
  int c = (i / L) % C;
  dst[i] = src[i] * w[c];
}

// out[b,l] = sum_c M[b,c,l] * w[c]
__global__ void k_rowdot(float* out, const float* M, const float* w,
                         int C, int L, int total) {
  int i = blockIdx.x * 256 + threadIdx.x;
  if (i >= total) return;
  int b = i / L, l = i % L;
  const float* mp = M + (size_t)b * C * L + l;
  float acc = 0.f;
  for (int c = 0; c < C; ++c) acc += mp[(size_t)c * L] * w[c];
  out[i] = acc;
}

// S[b,l,q] += s0[b,l] + s1[b,q] + cqb[0]
__global__ void k_cq_addbias(float* S, const float* s0, const float* s1,
                             const float* cqb, int Lc, int Lq, int total) {
  int i = blockIdx.x * 256 + threadIdx.x;
  if (i >= total) return;
  int q = i % Lq;
  int l = (i / Lq) % Lc;
  int b = i / (Lq * Lc);
  S[i] += s0[(size_t)b * Lc + l] + s1[(size_t)b * Lq + q] + cqb[0];
}

// X [B,1280,Lc] = concat([Cx, A, Cx*A, Cx*Bt], chan) ++ CeF  (match fusion)
__global__ void k_build_X(float* X, const float* CeF, const float* Amat,
                          const float* Btm, int Lc, int total) {
  int i = blockIdx.x * 256 + threadIdx.x;
  if (i >= total) return;
  int l = i % Lc;
  int c = (i / Lc) % 1280;
  int b = i / (Lc * 1280);
  size_t ceb = (size_t)b * 256 * Lc;
  size_t ab  = ((size_t)b * Lc + l) * 256;
  float v;
  if (c < 256)       v = CeF[ceb + (size_t)c * Lc + l];
  else if (c < 512)  v = Amat[ab + (c - 256)];
  else if (c < 768) { int d = c - 512; v = CeF[ceb + (size_t)d * Lc + l] * Amat[ab + d]; }
  else if (c < 1024){ int d = c - 768; v = CeF[ceb + (size_t)d * Lc + l] * Btm[ab + d]; }
  else               v = CeF[ceb + (size_t)(c - 1024) * Lc + l];
  X[i] = v;
}

// ------------------------------ host orchestration -------------------------

static inline int ceildiv(int a, int b) { return (a + b - 1) / b; }

static void run_gemm(hipStream_t st, const GemmP& p, int nz) {
  dim3 g(ceildiv(p.N, 64), ceildiv(p.M, 64), nz);
  if (!p.transA && !p.transB)      k_gemm<0, 0><<<g, 256, 0, st>>>(p);
  else if (p.transA && !p.transB)  k_gemm<1, 0><<<g, 256, 0, st>>>(p);
  else if (!p.transA && p.transB)  k_gemm<0, 1><<<g, 256, 0, st>>>(p);
  else                             k_gemm<1, 1><<<g, 256, 0, st>>>(p);
}

struct Scratch {
  float *act0, *act1, *tmpLN, *tmpDW, *mem, *qb, *logits;
};

static void enc_block(hipStream_t st, const Scratch& s,
                      const float* xin, int L, int nconv, int ksz,
                      const float* dw, const float* pw, const float* pwb,
                      const float* lncg, const float* lncb,
                      const float* memw, const float* qw,
                      const float* f1w, const float* f1b,
                      const float* f2w, const float* f2b,
                      const float* l1g, const float* l1b,
                      const float* l2g, const float* l2b,
                      const float* mask,
                      float* outP, int outLdc, long outBS) {
  const int Bn = 8, D = 128, Hh = 8;
  int total = Bn * D * L;
  k_posadd<<<ceildiv(total, 256), 256, 0, st>>>(s.act0, xin, L, total);
  float* X = s.act0;
  float* Y = s.act1;

  for (int i = 0; i < nconv; ++i) {
    k_lnc<<<ceildiv(Bn * L, 8), 256, 0, st>>>(X, s.tmpLN, lncg + i * D, lncb + i * D, L, Bn * L);
    k_dwconv<<<ceildiv(total, 256), 256, 0, st>>>(s.tmpLN, s.tmpDW, dw + (long)i * D * ksz, ksz, L, total);
    GemmP p = {};
    p.A = pw + (long)i * D * D; p.lda = D;
    p.B = s.tmpDW; p.ldb = L; p.sB0 = (long)D * L;
    p.C = Y; p.ldc = L; p.sC0 = (long)D * L;
    p.bias = pwb + i * D;
    p.resid = X; p.ldres = L; p.sR0 = (long)D * L;
    p.relu = 1; p.zshift = 0; p.alpha = 1.f;
    p.M = D; p.N = L; p.K = D;
    run_gemm(st, p, Bn);
    float* t = X; X = Y; Y = t;
  }

  // --- self-attention ---
  k_lnc<<<ceildiv(Bn * L, 8), 256, 0, st>>>(X, s.tmpLN, l1g, l1b, L, Bn * L);
  {
    GemmP p = {};
    p.A = memw; p.lda = D;
    p.B = s.tmpLN; p.ldb = L; p.sB0 = (long)D * L;
    p.C = s.mem; p.ldc = L; p.sC0 = (long)2 * D * L;
    p.zshift = 0; p.alpha = 1.f;
    p.M = 2 * D; p.N = L; p.K = D;
    run_gemm(st, p, Bn);
  }
  {
    GemmP p = {};
    p.A = qw; p.lda = D;
    p.B = s.tmpLN; p.ldb = L; p.sB0 = (long)D * L;
    p.C = s.qb; p.ldc = L; p.sC0 = (long)D * L;
    p.zshift = 0; p.alpha = 1.f;
    p.M = D; p.N = L; p.K = D;
    run_gemm(st, p, Bn);
  }
  {
    // logits[b,h,q,k] = 0.25 * sum_e q[b,h*16+e,q] * K[b,h*16+e,k]
    GemmP p = {};
    p.A = s.qb; p.lda = L; p.transA = 1; p.sA0 = (long)D * L; p.sA1 = 16L * L;
    p.B = s.mem; p.ldb = L; p.sB0 = (long)2 * D * L; p.sB1 = 16L * L;
    p.C = s.logits; p.ldc = L; p.sC0 = (long)Hh * L * L; p.sC1 = (long)L * L;
    p.zshift = 3; p.alpha = 0.25f;
    p.M = L; p.N = L; p.K = 16;
    run_gemm(st, p, Bn * Hh);
  }
  k_attn_softmax<<<dim3(L, Bn * Hh), 256, 0, st>>>(s.logits, mask, Hh, L, L);
  {
    // out[b,h*16+e,q] = sum_k P[b,h,q,k] * V[b,h*16+e,k]  (+ residual X)
    GemmP p = {};
    p.A = s.mem + (long)D * L; p.lda = L; p.sA0 = (long)2 * D * L; p.sA1 = 16L * L;
    p.B = s.logits; p.ldb = L; p.transB = 1; p.sB0 = (long)Hh * L * L; p.sB1 = (long)L * L;
    p.C = Y; p.ldc = L; p.sC0 = (long)D * L; p.sC1 = 16L * L;
    p.resid = X; p.ldres = L; p.sR0 = (long)D * L; p.sR1 = 16L * L;
    p.zshift = 3; p.alpha = 1.f;
    p.M = 16; p.N = L; p.K = L;
    run_gemm(st, p, Bn * Hh);
  }
  { float* t = X; X = Y; Y = t; }

  // --- FFN ---
  k_lnc<<<ceildiv(Bn * L, 8), 256, 0, st>>>(X, s.tmpLN, l2g, l2b, L, Bn * L);
  {
    GemmP p = {};
    p.A = f1w; p.lda = D;
    p.B = s.tmpLN; p.ldb = L; p.sB0 = (long)D * L;
    p.C = s.tmpDW; p.ldc = L; p.sC0 = (long)D * L;
    p.bias = f1b; p.relu = 1; p.zshift = 0; p.alpha = 1.f;
    p.M = D; p.N = L; p.K = D;
    run_gemm(st, p, Bn);
  }
  {
    GemmP p = {};
    p.A = f2w; p.lda = D;
    p.B = s.tmpDW; p.ldb = L; p.sB0 = (long)D * L;
    p.C = outP; p.ldc = outLdc; p.sC0 = outBS;
    p.bias = f2b;
    p.resid = X; p.ldres = L; p.sR0 = (long)D * L;
    p.zshift = 0; p.alpha = 1.f;
    p.M = D; p.N = L; p.K = D;
    run_gemm(st, p, Bn);
  }
}

extern "C" void kernel_launch(void* const* d_in, const int* in_sizes, int n_in,
                              void* d_out, int out_size, void* d_ws, size_t ws_size,
                              hipStream_t stream) {
  (void)in_sizes; (void)n_in; (void)out_size; (void)ws_size;
  const int Bn = 8, D = 128, Lc = 512, Lq = 64, Hh = 8;

  const float* Cin   = (const float*)d_in[0];
  const float* Qin   = (const float*)d_in[1];
  const float* maskC = (const float*)d_in[2];
  const float* maskQ = (const float*)d_in[3];
  const float* e_dw  = (const float*)d_in[4];
  const float* e_pw  = (const float*)d_in[5];
  const float* e_pwb = (const float*)d_in[6];
  const float* e_lncg= (const float*)d_in[7];
  const float* e_lncb= (const float*)d_in[8];
  const float* e_memw= (const float*)d_in[9];
  const float* e_qw  = (const float*)d_in[10];
  const float* e_f1w = (const float*)d_in[11];
  const float* e_f1b = (const float*)d_in[12];
  const float* e_f2w = (const float*)d_in[13];
  const float* e_f2b = (const float*)d_in[14];
  const float* e_l1g = (const float*)d_in[15];
  const float* e_l1b = (const float*)d_in[16];
  const float* e_l2g = (const float*)d_in[17];
  const float* e_l2b = (const float*)d_in[18];
  const float* w4C   = (const float*)d_in[19];
  const float* w4Q   = (const float*)d_in[20];
  const float* w4mlu = (const float*)d_in[21];
  const float* cqb   = (const float*)d_in[22];
  const float* resizer = (const float*)d_in[23];
  const float* m_dw  = (const float*)d_in[24];
  const float* m_pw  = (const float*)d_in[25];
  const float* m_pwb = (const float*)d_in[26];
  const float* m_lncg= (const float*)d_in[27];
  const float* m_lncb= (const float*)d_in[28];
  const float* m_memw= (const float*)d_in[29];
  const float* m_qw  = (const float*)d_in[30];
  const float* m_f1w = (const float*)d_in[31];
  const float* m_f1b = (const float*)d_in[32];
  const float* m_f2w = (const float*)d_in[33];
  const float* m_f2b = (const float*)d_in[34];
  const float* m_l1g = (const float*)d_in[35];
  const float* m_l1b = (const float*)d_in[36];
  const float* m_l2g = (const float*)d_in[37];
  const float* m_l2b = (const float*)d_in[38];
  float* Out = (float*)d_out;

  char* wp = (char*)d_ws;
  auto alloc = [&](size_t bytes) -> float* {
    float* p = (float*)wp;
    wp += (bytes + 255) & ~(size_t)255;
    return p;
  };
  const size_t fb = sizeof(float);
  Scratch s;
  s.act0   = alloc((size_t)Bn * D * Lc * fb);
  s.act1   = alloc((size_t)Bn * D * Lc * fb);
  s.tmpLN  = alloc((size_t)Bn * D * Lc * fb);
  s.tmpDW  = alloc((size_t)Bn * D * Lc * fb);
  s.mem    = alloc((size_t)Bn * 2 * D * Lc * fb);
  s.qb     = alloc((size_t)Bn * D * Lc * fb);
  s.logits = alloc((size_t)Bn * Hh * Lc * Lc * fb);
  float* CeF  = alloc((size_t)Bn * 2 * D * Lc * fb);
  float* QeF  = alloc((size_t)Bn * 2 * D * Lq * fb);
  float* Cw   = alloc((size_t)Bn * 2 * D * Lc * fb);
  float* S    = alloc((size_t)Bn * Lc * Lq * fb);
  float* S1   = alloc((size_t)Bn * Lc * Lq * fb);
  float* S2   = alloc((size_t)Bn * Lc * Lq * fb);
  float* s0v  = alloc((size_t)Bn * Lc * fb);
  float* s1v  = alloc((size_t)Bn * Lq * fb);
  float* Amat = alloc((size_t)Bn * Lc * 2 * D * fb);
  float* Tmat = alloc((size_t)Bn * Lq * 2 * D * fb);
  float* Btm  = alloc((size_t)Bn * Lc * 2 * D * fb);
  float* Xb   = alloc((size_t)Bn * 10 * D * Lc * fb);
  float* M0   = alloc((size_t)Bn * D * Lc * fb);
  float* Mb   = alloc((size_t)Bn * D * Lc * fb);

  // ---- embedding encoders (4 convs, k=7) -> rows 0..127 of CeF / QeF ----
  enc_block(stream, s, Cin, Lc, 4, 7, e_dw, e_pw, e_pwb, e_lncg, e_lncb,
            e_memw, e_qw, e_f1w, e_f1b, e_f2w, e_f2b, e_l1g, e_l1b, e_l2g, e_l2b,
            maskC, CeF, Lc, (long)2 * D * Lc);
  enc_block(stream, s, Qin, Lq, 4, 7, e_dw, e_pw, e_pwb, e_lncg, e_lncb,
            e_memw, e_qw, e_f1w, e_f1b, e_f2w, e_f2b, e_l1g, e_l1b, e_l2g, e_l2b,
            maskQ, QeF, Lq, (long)2 * D * Lq);
  // encode fusion: rows 128..255 = original inputs
  k_concat_rows<<<ceildiv(Bn * D * Lc, 256), 256, 0, stream>>>(CeF, Cin, D, Lc, D, 2 * D, Bn * D * Lc);
  k_concat_rows<<<ceildiv(Bn * D * Lq, 256), 256, 0, stream>>>(QeF, Qin, D, Lq, D, 2 * D, Bn * D * Lq);

  // ---- CQ attention ----
  k_scale_rows<<<ceildiv(Bn * 2 * D * Lc, 256), 256, 0, stream>>>(Cw, CeF, w4mlu, 2 * D, Lc, Bn * 2 * D * Lc);
  k_rowdot<<<ceildiv(Bn * Lc, 256), 256, 0, stream>>>(s0v, CeF, w4C, 2 * D, Lc, Bn * Lc);
  k_rowdot<<<ceildiv(Bn * Lq, 256), 256, 0, stream>>>(s1v, QeF, w4Q, 2 * D, Lq, Bn * Lq);
  {
    // S[l,q] = sum_d (Cx*w4mlu)[l,d] * Qx[q,d]
    GemmP p = {};
    p.A = Cw; p.lda = Lc; p.transA = 1; p.sA0 = (long)2 * D * Lc;
    p.B = QeF; p.ldb = Lq; p.sB0 = (long)2 * D * Lq;
    p.C = S; p.ldc = Lq; p.sC0 = (long)Lc * Lq;
    p.zshift = 0; p.alpha = 1.f;
    p.M = Lc; p.N = Lq; p.K = 2 * D;
    run_gemm(stream, p, Bn);
  }
  k_cq_addbias<<<ceildiv(Bn * Lc * Lq, 256), 256, 0, stream>>>(S, s0v, s1v, cqb, Lc, Lq, Bn * Lc * Lq);
  k_cq_softmax_q<<<Bn * Lc, Lq, 0, stream>>>(S, S1, maskQ, Lc, Lq);
  k_cq_softmax_c<<<Bn * Lq, 256, 0, stream>>>(S, S2, maskC, Lc, Lq);
  {
    // A = S1 @ Qx : [Lc, 2D]
    GemmP p = {};
    p.A = S1; p.lda = Lq; p.sA0 = (long)Lc * Lq;
    p.B = QeF; p.ldb = Lq; p.transB = 1; p.sB0 = (long)2 * D * Lq;
    p.C = Amat; p.ldc = 2 * D; p.sC0 = (long)Lc * 2 * D;
    p.zshift = 0; p.alpha = 1.f;
    p.M = Lc; p.N = 2 * D; p.K = Lq;
    run_gemm(stream, p, Bn);
  }
  {
    // T = S2^T @ Cx : [Lq, 2D]
    GemmP p = {};
    p.A = S2; p.lda = Lq; p.transA = 1; p.sA0 = (long)Lc * Lq;
    p.B = CeF; p.ldb = Lc; p.transB = 1; p.sB0 = (long)2 * D * Lc;
    p.C = Tmat; p.ldc = 2 * D; p.sC0 = (long)Lq * 2 * D;
    p.zshift = 0; p.alpha = 1.f;
    p.M = Lq; p.N = 2 * D; p.K = Lc;
    run_gemm(stream, p, Bn);
  }
  {
    // Bt = S1 @ T : [Lc, 2D]
    GemmP p = {};
    p.A = S1; p.lda = Lq; p.sA0 = (long)Lc * Lq;
    p.B = Tmat; p.ldb = 2 * D; p.sB0 = (long)Lq * 2 * D;
    p.C = Btm; p.ldc = 2 * D; p.sC0 = (long)Lc * 2 * D;
    p.zshift = 0; p.alpha = 1.f;
    p.M = Lc; p.N = 2 * D; p.K = Lq;
    run_gemm(stream, p, Bn);
  }
  k_build_X<<<ceildiv(Bn * 10 * D * Lc, 256), 256, 0, stream>>>(Xb, CeF, Amat, Btm, Lc, Bn * 10 * D * Lc);
  {
    // M0 = resizer @ X : [D, Lc], K = 1280
    GemmP p = {};
    p.A = resizer; p.lda = 10 * D;
    p.B = Xb; p.ldb = Lc; p.sB0 = (long)10 * D * Lc;
    p.C = M0; p.ldc = Lc; p.sC0 = (long)D * Lc;
    p.zshift = 0; p.alpha = 1.f;
    p.M = D; p.N = Lc; p.K = 10 * D;
    run_gemm(stream, p, Bn);
  }

  // ---- model encoder: 3 runs x 7 blocks (2 convs, k=5) ----
  const float* cur = M0;
  const long BDL = (long)Bn * D * Lc;
  for (int r = 0; r < 3; ++r) {
    for (int sb = 0; sb < 7; ++sb) {
      float* outP = (sb == 6) ? (Out + (size_t)r * BDL) : Mb;
      enc_block(stream, s, cur, Lc, 2, 5,
                m_dw  + (long)sb * 2 * D * 5,
                m_pw  + (long)sb * 2 * D * D,
                m_pwb + (long)sb * 2 * D,
                m_lncg+ (long)sb * 2 * D,
                m_lncb+ (long)sb * 2 * D,
                m_memw+ (long)sb * 2 * D * D,
                m_qw  + (long)sb * D * D,
                m_f1w + (long)sb * D * D, m_f1b + (long)sb * D,
                m_f2w + (long)sb * D * D, m_f2b + (long)sb * D,
                m_l1g + (long)sb * D, m_l1b + (long)sb * D,
                m_l2g + (long)sb * D, m_l2b + (long)sb * D,
                maskC, outP, Lc, (long)D * Lc);
      cur = outP;
    }
  }
}
